// TerrainAwareResBlock_29016799052466
// MI455X (gfx1250) — matrix operand order repaired and checked by reading, not compiled, not run on hardware
//
#include <hip/hip_runtime.h>

typedef __attribute__((ext_vector_type(16))) _Float16 v16h;
typedef __attribute__((ext_vector_type(8)))  float    v8f;
typedef __attribute__((ext_vector_type(4)))  unsigned uint4v;
typedef __attribute__((ext_vector_type(8)))  unsigned uint8v;

#define DEVINL static __device__ __forceinline__

DEVINL v8f wmma16(v16h a, v16h b, v8f c) {
  return __builtin_amdgcn_wmma_f32_16x16x32_f16(false, a, false, b, (short)0, c,
                                                false, false);
}

DEVINL v8f vzero8() {
  v8f z;
#pragma unroll
  for (int i = 0; i < 8; ++i) z[i] = 0.f;
  return z;
}

// LDS byte offset of a generic pointer to a __shared__ object:
// ISA 10.2: LDS aperture addresses map via addr[31:0].
DEVINL unsigned lds_offset(const void* p) { return (unsigned)(size_t)p; }

// CDNA5 async copy: 16 contiguous bytes global -> 16 contiguous bytes LDS,
// per active lane. Tracked by ASYNCcnt.
DEVINL void async_b128_to_lds(unsigned ldsoff, const void* gptr) {
  asm volatile("global_load_async_to_lds_b128 %0, %1, off"
               :: "v"(ldsoff), "v"(gptr)
               : "memory");
}
DEVINL void wait_asynccnt0() {
  asm volatile("s_wait_asynccnt 0x0" ::: "memory");
}

// CDNA5 Tensor Data Mover: 2-D tile load described by D# groups 0/1 (SGPRs).
DEVINL void tdm_load_2d(uint4v g0, uint8v g1) {
  asm volatile("tensor_load_to_lds %0, %1" :: "s"(g0), "s"(g1) : "memory");
}

// A fragment: 16x32 f16, row-major source with given stride.
// lane L: m = L&15 ; VGPR j holds K = (j>>2)*16 + (L>>4)*8 + (j&3)*2 (+1)
DEVINL v16h frag_a(const _Float16* p, int row0, int col0, int stride, int lane) {
  const _Float16* base = p + (row0 + (lane & 15)) * stride + col0;
  const int half8 = (lane >> 4) << 3;
  v16h f;
#pragma unroll
  for (int j = 0; j < 8; ++j) {
    int k = ((j >> 2) << 4) + half8 + ((j & 3) << 1);
    f[2 * j]     = base[k];
    f[2 * j + 1] = base[k + 1];
  }
  return f;
}

// B fragment: 32x16 f16 from a K-transposed [n][k] layout with given stride.
// lane L: n = L&15 ; element e holds K = (e>>3)*16 + (L>>4)*8 + (e&7)
DEVINL v16h frag_bT(const _Float16* pT, int col0, int k0, int stride, int lane) {
  const _Float16* base = pT + (col0 + (lane & 15)) * stride + k0 + ((lane >> 4) << 3);
  v16h f;
#pragma unroll
  for (int e = 0; e < 16; ++e) f[e] = base[((e >> 3) << 4) + (e & 7)];
  return f;
}

// ---------------------------------------------------------------- cast f32->f16
__global__ void cast_to_h(const float* __restrict__ src, _Float16* __restrict__ dst,
                          int n) {
  int i = blockIdx.x * blockDim.x + threadIdx.x;
  if (i < n) dst[i] = (_Float16)src[i];
}

// --------------------------------------------------- terrain projection t@W + b
// grid(B=4), block(256). t:[B][128], W:[128][256], out:[B][256]
__global__ void terrain_proj(const float* __restrict__ t, const float* __restrict__ W,
                             const float* __restrict__ bias, float* __restrict__ out) {
  const int b = blockIdx.x, c = threadIdx.x;
  const float* tb = t + b * 128;
  float acc = bias[c];
#pragma unroll 8
  for (int i = 0; i < 128; ++i) acc += tb[i] * W[i * 256 + c];
  out[b * 256 + c] = acc;
}

// ----------------------------------------------------------- conv 3x3 as GEMM
// X: [B*64*64][256] f16, Wt: [9][256][256] f16, Out: [16384][256] f32 (+bias)
// grid (16384/64, 256/128) = (256,2), block 256
__global__ __launch_bounds__(256) void conv3x3_wmma(
    const _Float16* __restrict__ X, const _Float16* __restrict__ Wt,
    const float* __restrict__ bias, float* __restrict__ Out) {
  __shared__ _Float16 As[64 * 40];
  __shared__ _Float16 Bt[128 * 40];
  const int mBase = blockIdx.x * 64;
  const int nBase = blockIdx.y * 128;
  const int tid = threadIdx.x;
  const int lane = tid & 31;
  const int wave = tid >> 5;
  const int wm = (wave >> 1) * 16;   // wave's m-tile
  const int wn = (wave & 1) * 64;    // wave's 4 n-tiles

  v8f acc[4];
#pragma unroll
  for (int t = 0; t < 4; ++t) acc[t] = vzero8();

  // A loader role: row r, 8-wide column segment c0 (16 bytes per lane)
  const int r  = tid >> 2;
  const int c0 = (tid & 3) * 8;
  const int p  = mBase + r;
  const int bb = p >> 12;
  const int hh = (p >> 6) & 63;
  const int ww = p & 63;
  // B loader role: k-row kk, 16-wide n segment n0
  const int kk = tid >> 3;
  const int n0 = (tid & 7) * 16;

  _Float16* dA = As + r * 40 + c0;
  const unsigned dA_lds = lds_offset(dA);

  for (int tap = 0; tap < 9; ++tap) {
    const int dy = tap / 3 - 1, dx = tap % 3 - 1;
    const int hs = hh + dy, wsd = ww + dx;
    const bool ok = ((unsigned)hs < 64u) && ((unsigned)wsd < 64u);
    const _Float16* srcA = X + ((((bb << 6) + hs) << 6) + wsd) * 256 + c0;
    for (int kc = 0; kc < 256; kc += 32) {
      if (ok) {
        // async 16B global -> LDS (ASYNCcnt-tracked)
        async_b128_to_lds(dA_lds, srcA + kc);
      } else {
#pragma unroll
        for (int i = 0; i < 8; ++i) dA[i] = (_Float16)0.f;
      }
      const _Float16* srcB = Wt + (tap * 256 + kc + kk) * 256 + nBase + n0;
#pragma unroll
      for (int i = 0; i < 16; ++i) Bt[(n0 + i) * 40 + kk] = srcB[i];
      wait_asynccnt0();
      __syncthreads();
      v16h af = frag_a(As, wm, 0, 40, lane);
#pragma unroll
      for (int t = 0; t < 4; ++t) {
        v16h bf = frag_bT(Bt, wn + t * 16, 0, 40, lane);
        acc[t] = wmma16(af, bf, acc[t]);
      }
      __syncthreads();
    }
  }
#pragma unroll
  for (int t = 0; t < 4; ++t) {
    const int n = nBase + wn + t * 16 + (lane & 15);
    const float bv = bias[n];
#pragma unroll
    for (int j = 0; j < 8; ++j) {
      const int m = mBase + wm + ((lane >> 4) << 3) + j;
      Out[m * 256 + n] = acc[t][j] + bv;
    }
  }
}

// ------------------------------------------------- generic GEMM (q/k/v proj)
// A:[16384][256] f16, Bw:[256][N] f16, Out:[16384][N] f16 (+bias). N<=128/blk.
// A tile is staged with the Tensor Data Mover (TDM): one 64x32 f16 2-D tile,
// LDS padding 4 dwords per 16 dwords -> padded stride 40 halves.
// grid (256, ceil(N/128)), block 256
__global__ __launch_bounds__(256) void gemm_qkv(
    const _Float16* __restrict__ A, const _Float16* __restrict__ Bw,
    const float* __restrict__ bias, int N, _Float16* __restrict__ Out) {
  __shared__ _Float16 As[64 * 40];
  __shared__ _Float16 Bt[128 * 40];
  const int mBase = blockIdx.x * 64;
  const int nBase = blockIdx.y * 128;
  const int tid = threadIdx.x;
  const int lane = tid & 31;
  const int wave = tid >> 5;
  const int wm = (wave >> 1) * 16;
  const int wn = (wave & 1) * 64;

  v8f acc[4];
#pragma unroll
  for (int t = 0; t < 4; ++t) acc[t] = vzero8();

  const int kk = tid >> 3;
  const int n0 = (tid & 7) * 16;

  const unsigned as_lds = lds_offset(&As[0]);
  const unsigned long long gbase =
      (unsigned long long)(size_t)(A + (size_t)mBase * 256);

  for (int kc = 0; kc < 256; kc += 32) {
    if (wave == 0) {
      // Build D# (group0 128b, group1 256b) for a 64x32 f16 tile of a
      // 16384x256 tensor; pad LDS rows by 4 dwords every 16 dwords.
      const unsigned long long ga = gbase + (unsigned)(kc * 2);
      uint4v g0;
      g0[0] = 1u;                                   // count=1, load descriptor
      g0[1] = as_lds;                               // lds_addr (bytes)
      g0[2] = (unsigned)ga;                         // global_addr[31:0]
      g0[3] = (unsigned)(ga >> 32) | 0x80000000u;   // addr[56:32] | type=2
      uint8v g1;
      g1[0] = (1u << 16)        // data_size = 2 bytes
            | (1u << 20)        // pad_enable
            | (3u << 22)        // pad_interval: 16 dwords
            | (3u << 25);       // pad_amount: 4 dwords
      g1[1] = 256u << 16;       // tensor_dim0 = 256 (elements)
      g1[2] = 16384u << 16;     // tensor_dim1 = 16384 (low 16 bits)
      g1[3] = 32u << 16;        // tile_dim0 = 32
      g1[4] = 64u;              // tile_dim1 = 64
      g1[5] = 256u;             // tensor_dim0_stride = 256
      g1[6] = 0u;
      g1[7] = 0u;
      tdm_load_2d(g0, g1);
    }
#pragma unroll
    for (int i = 0; i < 16; ++i) {
      int n = nBase + n0 + i;
      Bt[(n0 + i) * 40 + kk] = (n < N) ? Bw[(kc + kk) * N + n] : (_Float16)0.f;
    }
    if (wave == 0) __builtin_amdgcn_s_wait_tensorcnt(0);
    __syncthreads();
    v16h af = frag_a(As, wm, 0, 40, lane);
#pragma unroll
    for (int t = 0; t < 4; ++t) {
      v16h bf = frag_bT(Bt, wn + t * 16, 0, 40, lane);
      acc[t] = wmma16(af, bf, acc[t]);
    }
    __syncthreads();
  }
#pragma unroll
  for (int t = 0; t < 4; ++t) {
    const int n = nBase + wn + t * 16 + (lane & 15);
    if (n < N) {
      const float bv = bias[n];
#pragma unroll
      for (int j = 0; j < 8; ++j) {
        const int m = mBase + wm + ((lane >> 4) << 3) + j;
        Out[m * N + n] = (_Float16)(acc[t][j] + bv);
      }
    }
  }
}

// -------------------------------------- TerrainAdaptiveNormalization (fused)
// grid(16384) block(256): one pixel per block, one channel per thread.
__global__ __launch_bounds__(256) void tan_norm(
    const float* __restrict__ In, const float* __restrict__ bn_g,
    const float* __restrict__ bn_b, const float* __restrict__ bn_m,
    const float* __restrict__ bn_v, const float* __restrict__ ln_g,
    const float* __restrict__ ln_b, const float* __restrict__ tsc,
    const float* __restrict__ tbi, const float* __restrict__ tcs,
    const float* __restrict__ tcb, const float* __restrict__ tf,
    _Float16* __restrict__ out_h, float* __restrict__ out_f, int do_relu) {
  const int p = blockIdx.x;
  const int c = threadIdx.x;
  const int b = p >> 12;
  const float eps = 1e-3f;
  float v = In[p * 256 + c];
  float n = (v - bn_m[c]) * rsqrtf(bn_v[c] + eps) * bn_g[c] + bn_b[c];

  __shared__ float redA[8], redB[8];
  float s1 = n, s2 = n * n;
#pragma unroll
  for (int o = 16; o > 0; o >>= 1) {
    s1 += __shfl_xor(s1, o, 32);
    s2 += __shfl_xor(s2, o, 32);
  }
  if ((c & 31) == 0) { redA[c >> 5] = s1; redB[c >> 5] = s2; }
  __syncthreads();
  float mu = 0.f, m2 = 0.f;
#pragma unroll
  for (int i = 0; i < 8; ++i) { mu += redA[i]; m2 += redB[i]; }
  mu *= (1.f / 256.f);
  m2 *= (1.f / 256.f);
  float var = m2 - mu * mu;
  float cn = (n - mu) * rsqrtf(var + eps) * ln_g[c] + ln_b[c];

  const int bc = b * 256 + c;
  float out = (n * (1.f + tsc[bc]) + tbi[bc] + cn * (1.f + tcs[bc]) + tcb[bc]) * 0.5f;
  if (do_relu) out = fmaxf(out, 0.f);
  if (tf) out += tf[bc];
  out_h[p * 256 + c] = (_Float16)out;
  if (out_f) out_f[p * 256 + c] = out;
}

// ------------------------------------------------ flash attention + epilogue
// grid (4096/64, B=4), block 256. Q,K:[B*4096][32] f16, V:[B*4096][256] f16
// Out = relu(gamma * softmax(QK^T/sqrt(32)) V + xa + x)
__global__ __launch_bounds__(256) void flash_attn(
    const _Float16* __restrict__ Q, const _Float16* __restrict__ K,
    const _Float16* __restrict__ V, const float* __restrict__ xa,
    const float* __restrict__ xin, const float* __restrict__ gamma_p,
    float* __restrict__ Out) {
  __shared__ float    Sbuf[64 * 68];      // 17408 B
  __shared__ _Float16 Pbuf[64 * 72];      //  9216 B
  __shared__ _Float16 Vt[256 * 72];       // 36864 B  (V tile transposed [n][k])
  __shared__ float    rowScale[64];
  __shared__ float    rowSumL[64];

  const int b = blockIdx.y;
  const int qBase = blockIdx.x * 64;
  const int tid = threadIdx.x;
  const int lane = tid & 31;
  const int wave = tid >> 5;
  const int wm  = (wave >> 1) * 16;   // m-tile (shared by S and O)
  const int wnS = (wave & 1) * 32;    // S: 2 n-tiles
  const int wnO = (wave & 1) * 128;   // O: 8 n-tiles
  const int half8 = (lane >> 4) << 3;

  const _Float16* Qb = Q + (size_t)b * 4096 * 32;
  const _Float16* Kb = K + (size_t)b * 4096 * 32;
  const _Float16* Vb = V + (size_t)b * 4096 * 256;

  // Q fragment for this wave's m-tile (c8 = 32 = exactly one WMMA-K)
  v16h qf = frag_a(Qb, qBase + wm, 0, 32, lane);

  v8f oacc[8];
#pragma unroll
  for (int t = 0; t < 8; ++t) oacc[t] = vzero8();

  // online-softmax state: 4 threads per row, row = tid>>2
  const int myRow = tid >> 2;
  const int myCol = (tid & 3) * 16;
  float mrow = -1e30f, lrow = 0.f;
  const float rscale = 0.17677669529663687f;  // 1/sqrt(32)

  for (int kb = 0; kb < 4096; kb += 64) {
    // stage V tile transposed: Vt[n][k]
    {
      const int kk2 = tid >> 2;          // 0..63
      const int n0  = (tid & 3) * 64;
      const _Float16* src = Vb + (size_t)(kb + kk2) * 256 + n0;
#pragma unroll
      for (int i = 0; i < 64; ++i) Vt[(n0 + i) * 72 + kk2] = src[i];
    }
    // S = Q K^T for this wave's two 16x16 tiles, K frags direct from global
    v8f sacc[2];
#pragma unroll
    for (int t = 0; t < 2; ++t) {
      sacc[t] = vzero8();
      v16h kf = frag_bT(Kb, kb + wnS + t * 16, 0, 32, lane);
      sacc[t] = wmma16(qf, kf, sacc[t]);
    }
#pragma unroll
    for (int t = 0; t < 2; ++t) {
      const int n = wnS + t * 16 + (lane & 15);
#pragma unroll
      for (int j = 0; j < 8; ++j) Sbuf[(wm + half8 + j) * 68 + n] = sacc[t][j] * rscale;
    }
    __syncthreads();
    // online softmax: each thread owns 16 columns of its row
    {
      const float* srow = Sbuf + myRow * 68 + myCol;
      float sv[16];
      float lmax = -1e30f;
#pragma unroll
      for (int i = 0; i < 16; ++i) { sv[i] = srow[i]; lmax = fmaxf(lmax, sv[i]); }
      lmax = fmaxf(lmax, __shfl_xor(lmax, 1, 32));
      lmax = fmaxf(lmax, __shfl_xor(lmax, 2, 32));
      const float mnew = fmaxf(mrow, lmax);
      const float sc = __expf(mrow - mnew);
      float ps = 0.f;
      _Float16* prow = Pbuf + myRow * 72 + myCol;
#pragma unroll
      for (int i = 0; i < 16; ++i) {
        float e = __expf(sv[i] - mnew);
        ps += e;
        prow[i] = (_Float16)e;
      }
      ps += __shfl_xor(ps, 1, 32);
      ps += __shfl_xor(ps, 2, 32);
      lrow = lrow * sc + ps;
      mrow = mnew;
      if ((tid & 3) == 0) rowScale[myRow] = sc;
    }
    __syncthreads();
    // rescale O accumulators by per-row exp(m_old - m_new)
    {
      float rs[8];
#pragma unroll
      for (int j = 0; j < 8; ++j) rs[j] = rowScale[wm + half8 + j];
#pragma unroll
      for (int t = 0; t < 8; ++t)
#pragma unroll
        for (int j = 0; j < 8; ++j) oacc[t][j] *= rs[j];
    }
    // O += P @ V  (K = 64 -> two WMMA-K chunks)
#pragma unroll
    for (int kc = 0; kc < 2; ++kc) {
      v16h pf = frag_a(Pbuf, wm, kc * 32, 72, lane);
#pragma unroll
      for (int t = 0; t < 8; ++t) {
        v16h vf = frag_bT(Vt, wnO + t * 16, kc * 32, 72, lane);
        oacc[t] = wmma16(pf, vf, oacc[t]);
      }
    }
    __syncthreads();
  }

  if ((tid & 3) == 0) rowSumL[myRow] = lrow;
  __syncthreads();
  const float gamma = gamma_p[0];
  float linv[8];
#pragma unroll
  for (int j = 0; j < 8; ++j) linv[j] = 1.f / rowSumL[wm + half8 + j];
#pragma unroll
  for (int t = 0; t < 8; ++t) {
    const int n = wnO + t * 16 + (lane & 15);
#pragma unroll
    for (int j = 0; j < 8; ++j) {
      const int m = qBase + wm + half8 + j;
      const size_t idx = ((size_t)b << 20) + (size_t)m * 256 + n;
      float val = oacc[t][j] * linv[j] * gamma + xa[idx] + xin[idx];
      Out[idx] = fmaxf(val, 0.f);
    }
  }
}

// ----------------------------------------------------------------- launcher
extern "C" void kernel_launch(void* const* d_in, const int* in_sizes, int n_in,
                              void* d_out, int out_size, void* d_ws, size_t ws_size,
                              hipStream_t stream) {
  (void)in_sizes; (void)n_in; (void)out_size; (void)ws_size;
  const float* x        = (const float*)d_in[0];
  const float* terr     = (const float*)d_in[1];
  const float* conv1_w  = (const float*)d_in[2];
  const float* conv1_b  = (const float*)d_in[3];
  const float* conv2_w  = (const float*)d_in[4];
  const float* conv2_b  = (const float*)d_in[5];
  const float* att_tp_w = (const float*)d_in[6];
  const float* att_tp_b = (const float*)d_in[7];
  const float* att_q_w  = (const float*)d_in[8];
  const float* att_q_b  = (const float*)d_in[9];
  const float* att_k_w  = (const float*)d_in[10];
  const float* att_k_b  = (const float*)d_in[11];
  const float* att_v_w  = (const float*)d_in[12];
  const float* att_v_b  = (const float*)d_in[13];
  const float* att_gamma= (const float*)d_in[14];
  const float* n1_bn_g = (const float*)d_in[15];
  const float* n1_bn_b = (const float*)d_in[16];
  const float* n1_bn_m = (const float*)d_in[17];
  const float* n1_bn_v = (const float*)d_in[18];
  const float* n1_ts_w = (const float*)d_in[19];
  const float* n1_ts_b = (const float*)d_in[20];
  const float* n1_tb_w = (const float*)d_in[21];
  const float* n1_tb_b = (const float*)d_in[22];
  const float* n1_ln_g = (const float*)d_in[23];
  const float* n1_ln_b = (const float*)d_in[24];
  const float* n1_cs_w = (const float*)d_in[25];
  const float* n1_cs_b = (const float*)d_in[26];
  const float* n1_cb_w = (const float*)d_in[27];
  const float* n1_cb_b = (const float*)d_in[28];
  const float* n2_bn_g = (const float*)d_in[29];
  const float* n2_bn_b = (const float*)d_in[30];
  const float* n2_bn_m = (const float*)d_in[31];
  const float* n2_bn_v = (const float*)d_in[32];
  const float* n2_ts_w = (const float*)d_in[33];
  const float* n2_ts_b = (const float*)d_in[34];
  const float* n2_tb_w = (const float*)d_in[35];
  const float* n2_tb_b = (const float*)d_in[36];
  const float* n2_ln_g = (const float*)d_in[37];
  const float* n2_ln_b = (const float*)d_in[38];
  const float* n2_cs_w = (const float*)d_in[39];
  const float* n2_cs_b = (const float*)d_in[40];
  const float* n2_cb_w = (const float*)d_in[41];
  const float* n2_cb_b = (const float*)d_in[42];

  const int M = 4 * 64 * 64;          // 16384 pixels
  char* ws = (char*)d_ws;
  size_t off = 0;
  auto alloc = [&](size_t bytes) -> char* {
    char* p = ws + off;
    off += (bytes + 255) & ~(size_t)255;
    return p;
  };
  float*    TVEC    = (float*)alloc(9 * 4 * 256 * sizeof(float));
  float*    CONVOUT = (float*)alloc((size_t)M * 256 * sizeof(float));
  float*    XA32    = (float*)alloc((size_t)M * 256 * sizeof(float));
  _Float16* W1H     = (_Float16*)alloc((size_t)9 * 256 * 256 * 2);
  _Float16* W2H     = (_Float16*)alloc((size_t)9 * 256 * 256 * 2);
  _Float16* QWH     = (_Float16*)alloc(256 * 32 * 2);
  _Float16* KWH     = (_Float16*)alloc(256 * 32 * 2);
  _Float16* VWH     = (_Float16*)alloc(256 * 256 * 2);
  _Float16* XH      = (_Float16*)alloc((size_t)M * 256 * 2);
  _Float16* H1H     = (_Float16*)alloc((size_t)M * 256 * 2);
  _Float16* XAH     = (_Float16*)alloc((size_t)M * 256 * 2);
  _Float16* QB      = (_Float16*)alloc((size_t)M * 32 * 2);
  _Float16* KB      = (_Float16*)alloc((size_t)M * 32 * 2);
  _Float16* VB      = (_Float16*)alloc((size_t)M * 256 * 2);

  // --- casts to f16
  const int nx = M * 256;
  cast_to_h<<<(nx + 255) / 256, 256, 0, stream>>>(x, XH, nx);
  cast_to_h<<<(9 * 256 * 256 + 255) / 256, 256, 0, stream>>>(conv1_w, W1H, 9 * 256 * 256);
  cast_to_h<<<(9 * 256 * 256 + 255) / 256, 256, 0, stream>>>(conv2_w, W2H, 9 * 256 * 256);
  cast_to_h<<<(256 * 32 + 255) / 256, 256, 0, stream>>>(att_q_w, QWH, 256 * 32);
  cast_to_h<<<(256 * 32 + 255) / 256, 256, 0, stream>>>(att_k_w, KWH, 256 * 32);
  cast_to_h<<<(256 * 256 + 255) / 256, 256, 0, stream>>>(att_v_w, VWH, 256 * 256);

  // --- terrain projections (9 vectors of [B][256])
  terrain_proj<<<4, 256, 0, stream>>>(terr, n1_ts_w, n1_ts_b, TVEC + 0 * 1024);
  terrain_proj<<<4, 256, 0, stream>>>(terr, n1_tb_w, n1_tb_b, TVEC + 1 * 1024);
  terrain_proj<<<4, 256, 0, stream>>>(terr, n1_cs_w, n1_cs_b, TVEC + 2 * 1024);
  terrain_proj<<<4, 256, 0, stream>>>(terr, n1_cb_w, n1_cb_b, TVEC + 3 * 1024);
  terrain_proj<<<4, 256, 0, stream>>>(terr, n2_ts_w, n2_ts_b, TVEC + 4 * 1024);
  terrain_proj<<<4, 256, 0, stream>>>(terr, n2_tb_w, n2_tb_b, TVEC + 5 * 1024);
  terrain_proj<<<4, 256, 0, stream>>>(terr, n2_cs_w, n2_cs_b, TVEC + 6 * 1024);
  terrain_proj<<<4, 256, 0, stream>>>(terr, n2_cb_w, n2_cb_b, TVEC + 7 * 1024);
  terrain_proj<<<4, 256, 0, stream>>>(terr, att_tp_w, att_tp_b, TVEC + 8 * 1024);

  // --- conv1 + TAN1 + relu
  conv3x3_wmma<<<dim3(M / 64, 2), 256, 0, stream>>>(XH, W1H, conv1_b, CONVOUT);
  tan_norm<<<M, 256, 0, stream>>>(CONVOUT, n1_bn_g, n1_bn_b, n1_bn_m, n1_bn_v,
                                  n1_ln_g, n1_ln_b, TVEC + 0 * 1024, TVEC + 1 * 1024,
                                  TVEC + 2 * 1024, TVEC + 3 * 1024, (const float*)nullptr,
                                  H1H, (float*)nullptr, 1);

  // --- conv2 + TAN2 (+ terrain injection -> xa)
  conv3x3_wmma<<<dim3(M / 64, 2), 256, 0, stream>>>(H1H, W2H, conv2_b, CONVOUT);
  tan_norm<<<M, 256, 0, stream>>>(CONVOUT, n2_bn_g, n2_bn_b, n2_bn_m, n2_bn_v,
                                  n2_ln_g, n2_ln_b, TVEC + 4 * 1024, TVEC + 5 * 1024,
                                  TVEC + 6 * 1024, TVEC + 7 * 1024, TVEC + 8 * 1024,
                                  XAH, XA32, 0);

  // --- q/k/v projections (TDM-staged A tiles)
  gemm_qkv<<<dim3(M / 64, 1), 256, 0, stream>>>(XAH, QWH, att_q_b, 32, QB);
  gemm_qkv<<<dim3(M / 64, 1), 256, 0, stream>>>(XAH, KWH, att_k_b, 32, KB);
  gemm_qkv<<<dim3(M / 64, 2), 256, 0, stream>>>(XAH, VWH, att_v_b, 256, VB);

  // --- flash attention + fused epilogue (gamma*O + xa + x, relu) -> d_out
  flash_attn<<<dim3(64, 4), 256, 0, stream>>>(QB, KB, VB, XA32, x, att_gamma,
                                              (float*)d_out);
}
